// LoFTREncoderLayer_70274254897271
// MI455X (gfx1250) — matrix operand order repaired and checked by reading, not compiled, not run on hardware
//
#include <hip/hip_runtime.h>

// ---------------- problem constants ----------------
#define B_   16
#define L_   4800
#define S_   4800
#define D_   256
#define H_   8
#define HD_  32
#define M_   (B_ * L_)      // 76800 rows
#define D2_  (2 * D_)       // 512
#define EPS_ATT 1e-6f
#define EPS_LN  1e-5f
#define KV_SPLIT 15          // 4800 = 15 slices * 320 rows (320 = 5 chunks of 64)

// ---------------- WMMA types ----------------
typedef __attribute__((ext_vector_type(16))) __bf16 v16bf;
typedef __attribute__((ext_vector_type(8)))  __bf16 v8bf;
typedef __attribute__((ext_vector_type(8)))  float  v8f;

__device__ __forceinline__ float  bf2f(__bf16 x) { return (float)x; }
__device__ __forceinline__ __bf16 f2bf(float x)  { return (__bf16)x; }

// act codes
#define ACT_F32   0   // raw f32 output (feeds LayerNorm)
#define ACT_ELU1  1   // elu(x)+1 -> bf16
#define ACT_SCALE 2   // x*sarg   -> bf16
#define ACT_RELU  3   // relu(x)  -> bf16

// ---------------- fp32 -> bf16 convert ----------------
__global__ void cvt_f32_bf16(const float* __restrict__ in, __bf16* __restrict__ out, long n) {
  long i = (long)blockIdx.x * blockDim.x + threadIdx.x;
  long stride = (long)gridDim.x * blockDim.x;
  for (; i < n; i += stride) out[i] = f2bf(in[i]);
}

// ---------------- generic NT GEMM: out = act(A[M,K] @ W[N,K]^T) ----------------
// Block tile 64x256, 8 waves (2x4), wave tile 32x64 = 2x4 WMMA 16x16 tiles.
// This shape reads the streaming A operand from HBM exactly once for N=256
// (weights are L2-resident). A operand may be split at column `ksplit`
// between A0/A1 so concat([query, message]) is never materialized.
__global__ __launch_bounds__(256)
void gemm_bf16_act(const __bf16* __restrict__ A0, const __bf16* __restrict__ A1,
                   int ksplit, int lda,
                   const __bf16* __restrict__ W, int ldw, int K,
                   float* __restrict__ outF, __bf16* __restrict__ outB,
                   int ldo, int act, float sarg)
{
  const int lane = threadIdx.x & 31;
  const int wave = threadIdx.x >> 5;      // 0..7
  const int wm   = wave >> 2;             // 0..1 (rows of waves)
  const int wn   = wave & 3;              // 0..3 (cols of waves)
  const long mBase = (long)blockIdx.x * 64  + wm * 32;
  const long nBase = (long)blockIdx.y * 256 + wn * 64;
  const int lr = lane & 15;               // index within 16
  const int lh = lane >> 4;               // half-wave select

  v8f acc[2][4];
#pragma unroll
  for (int s = 0; s < 2; ++s)
#pragma unroll
    for (int t = 0; t < 4; ++t)
#pragma unroll
      for (int r = 0; r < 8; ++r) acc[s][t][r] = 0.f;

  for (int k = 0; k < K; k += 32) {
    const __bf16* abase; int acol;
    if (k < ksplit) { abase = A0; acol = k; }
    else            { abase = A1; acol = k - ksplit; }

    // A fragment (16x32 bf16): ISA layout -> two contiguous B128 loads per lane.
    // lane m = lane%16 ; VGPR0-3 hold K = (lane<16?0:8)+0..7 ; VGPR4-7 hold +16.
    v16bf afrag[2];
#pragma unroll
    for (int s = 0; s < 2; ++s) {
      const __bf16* ap = abase + (mBase + s * 16 + lr) * (long)lda + acol;
      const v8bf lo = *(const v8bf*)(ap + lh * 8);
      const v8bf hi = *(const v8bf*)(ap + 16 + lh * 8);
#pragma unroll
      for (int i = 0; i < 8; ++i) { afrag[s][i] = lo[i]; afrag[s][8 + i] = hi[i]; }
      // prefetch next k-tile of the streaming A operand (global_prefetch_b8)
      if (k + 32 < K) __builtin_prefetch(ap + 32, 0, 0);
    }

    // B fragment (32x16 bf16, from row-major W[N,K]): lane n = lane%16,
    // same K layout as A: elements 0-7 -> K = 8*(lane/16)+0..7 ; 8-15 -> +16.
    v16bf bfrag[4];
#pragma unroll
    for (int t = 0; t < 4; ++t) {
      const __bf16* bp = W + (nBase + t * 16 + lr) * (long)ldw + k + lh * 8;
      const v8bf lo = *(const v8bf*)(bp);
      const v8bf hi = *(const v8bf*)(bp + 16);
#pragma unroll
      for (int i = 0; i < 8; ++i) { bfrag[t][i] = lo[i]; bfrag[t][8 + i] = hi[i]; }
    }

#pragma unroll
    for (int s = 0; s < 2; ++s)
#pragma unroll
      for (int t = 0; t < 4; ++t)
        acc[s][t] = __builtin_amdgcn_wmma_f32_16x16x32_bf16(
            false, afrag[s], false, bfrag[t], (short)0, acc[s][t], false, false);
  }

  // epilogue: C/D layout -> m = r + 8*(lane/16), n = lane%16
#pragma unroll
  for (int s = 0; s < 2; ++s)
#pragma unroll
    for (int t = 0; t < 4; ++t)
#pragma unroll
      for (int r = 0; r < 8; ++r) {
        long m = mBase + s * 16 + r + lh * 8;
        long n = nBase + t * 16 + lr;
        float x = acc[s][t][r];
        if (act == ACT_F32)        outF[m * (long)ldo + n] = x;
        else if (act == ACT_ELU1)  outB[m * (long)ldo + n] = f2bf(x > 0.f ? x + 1.f : __expf(x));
        else if (act == ACT_SCALE) outB[m * (long)ldo + n] = f2bf(x * sarg);
        else                       outB[m * (long)ldo + n] = f2bf(x > 0.f ? x : 0.f);
      }
}

// ---------------- KV[b,h,d,v] += sum_s Kf[b,s,h,d]*Vs[b,s,h,v]; Ksum[b,h,d] ----------------
// grid = (B*H, KV_SPLIT); each block reduces a 320-row slice of s and
// atomically accumulates its 32x32 partial into KV (zero-initialized).
__global__ __launch_bounds__(256)
void kv_ksum_kernel(const __bf16* __restrict__ Kf, const __bf16* __restrict__ Vs,
                    float* __restrict__ KV, float* __restrict__ Ksum)
{
  const int bh = blockIdx.x;                     // 0..B*H-1
  const int b  = bh / H_;
  const int h  = bh % H_;
  const int sBeg = blockIdx.y * (S_ / KV_SPLIT); // slices of 320 rows
  const int sEnd = sBeg + (S_ / KV_SPLIT);
  const __bf16* kbase = Kf + (long)b * S_ * D_ + h * HD_;
  const __bf16* vbase = Vs + (long)b * S_ * D_ + h * HD_;

  __shared__ __bf16 kch[64][32];
  __shared__ __bf16 vch[64][32];

  const int t = threadIdx.x;
  const int v = t & 31;        // output column
  const int g = t >> 5;        // output row group (0..7)
  float acc[4] = {0.f, 0.f, 0.f, 0.f};
  float ks = 0.f;

  const int row = t >> 2;            // 0..63, one v8bf per thread per matrix
  const int col = (t & 3) * 8;

  for (int s0 = sBeg; s0 < sEnd; s0 += 64) {
    *(v8bf*)&kch[row][col] = *(const v8bf*)(kbase + (long)(s0 + row) * D_ + col);
    *(v8bf*)&vch[row][col] = *(const v8bf*)(vbase + (long)(s0 + row) * D_ + col);
    __syncthreads();
#pragma unroll 4
    for (int ss = 0; ss < 64; ++ss) {
      float vv = bf2f(vch[ss][v]);
#pragma unroll
      for (int i = 0; i < 4; ++i) acc[i] += bf2f(kch[ss][g + i * 8]) * vv;
      if (t < 32) ks += bf2f(kch[ss][t]);
    }
    __syncthreads();
  }

  float* kvout = KV + (long)bh * HD_ * HD_;
#pragma unroll
  for (int i = 0; i < 4; ++i) atomicAdd(&kvout[(g + i * 8) * HD_ + v], acc[i]);
  if (t < 32) atomicAdd(&Ksum[(long)bh * HD_ + t], ks);
}

// ---------------- message[b,l,h,v] = (Qf . KV) * S / (Qf . Ksum + eps) ----------------
__global__ __launch_bounds__(256)
void message_kernel(const __bf16* __restrict__ Qf, const float* __restrict__ KV,
                    const float* __restrict__ Ksum, __bf16* __restrict__ msg)
{
  const long bl = blockIdx.x;                // row index in [0, B*L)
  const int  b  = (int)(bl / L_);
  const int  t  = threadIdx.x;               // 0..255
  const int  h  = t >> 5, v = t & 31;

  __shared__ float q[D_];
  q[t] = bf2f(Qf[bl * D_ + t]);
  __syncthreads();

  const float* kv   = KV   + ((long)b * H_ + h) * HD_ * HD_;
  const float* ksum = Ksum + ((long)b * H_ + h) * HD_;
  float dotv = 0.f, z = 0.f;
#pragma unroll
  for (int d = 0; d < HD_; ++d) {
    float qd = q[h * HD_ + d];
    dotv += qd * kv[d * HD_ + v];
    z    += qd * ksum[d];
  }
  msg[bl * D_ + t] = f2bf(dotv * (1.f / (z + EPS_ATT)) * (float)S_);
}

// ---------------- LayerNorm over D=256; optional bf16 out, optional residual f32 out ----------------
__global__ __launch_bounds__(256)
void ln_kernel(const float* __restrict__ in, const float* __restrict__ gam,
               const float* __restrict__ bet, const float* __restrict__ resid,
               float* __restrict__ outF, __bf16* __restrict__ outB)
{
  const long row = blockIdx.x;
  const int  t   = threadIdx.x;
  float x = in[row * (long)D_ + t];

  __shared__ float red[256];
  red[t] = x; __syncthreads();
#pragma unroll
  for (int off = 128; off > 0; off >>= 1) { if (t < off) red[t] += red[t + off]; __syncthreads(); }
  float mu = red[0] * (1.f / D_);
  __syncthreads();

  float d0 = x - mu;
  red[t] = d0 * d0; __syncthreads();
#pragma unroll
  for (int off = 128; off > 0; off >>= 1) { if (t < off) red[t] += red[t + off]; __syncthreads(); }
  float var = red[0] * (1.f / D_);

  float y = d0 * rsqrtf(var + EPS_LN) * gam[t] + bet[t];
  if (outB) outB[row * (long)D_ + t] = f2bf(y);
  if (outF) outF[row * (long)D_ + t] = resid[row * (long)D_ + t] + y;
}

// ---------------- launcher ----------------
extern "C" void kernel_launch(void* const* d_in, const int* in_sizes, int n_in,
                              void* d_out, int out_size, void* d_ws, size_t ws_size,
                              hipStream_t stream) {
  (void)in_sizes; (void)n_in; (void)out_size; (void)ws_size;
  const float* query   = (const float*)d_in[0];
  const float* context = (const float*)d_in[1];
  const float* Wq = (const float*)d_in[2];
  const float* Wk = (const float*)d_in[3];
  const float* Wv = (const float*)d_in[4];
  const float* Wm = (const float*)d_in[5];
  const float* W1 = (const float*)d_in[6];
  const float* W2 = (const float*)d_in[7];
  const float* g1 = (const float*)d_in[8];
  const float* b1 = (const float*)d_in[9];
  const float* g2 = (const float*)d_in[10];
  const float* b2 = (const float*)d_in[11];

  // workspace layout (bytes); MD2 = bf16 [M,D] plane
  const size_t MD2 = (size_t)M_ * D_ * 2;   // 39.3 MB
  const size_t MD4 = (size_t)M_ * D_ * 4;   // 78.6 MB
  char* w = (char*)d_ws;
  __bf16* qbf    = (__bf16*)(w + 0 * MD2);        // query bf16 (lives through MLP1)
  __bf16* cbf    = (__bf16*)(w + 1 * MD2);        // context bf16; reused later as msg
  __bf16* Qf     = (__bf16*)(w + 2 * MD2);
  __bf16* Kf     = (__bf16*)(w + 3 * MD2);        // dead after KV; reused as msg_ln
  __bf16* Vs     = (__bf16*)(w + 4 * MD2);        // dead after KV; start of mid region
  float*  mergeF = (float*) (w + 5 * MD2);        // dead after LN1 (overlapped by mid)
  __bf16* msg    = cbf;                            // reuse
  __bf16* msg_ln = Kf;                             // reuse
  __bf16* mid    = Vs;                             // [M, 2D] bf16 over Vs+mergeF regions
  float*  o2     = (float*) (w + 7 * MD2);        // MLP2 f32 output
  char*   wp     = w + 7 * MD2 + MD4;
  __bf16* wqb = (__bf16*)(wp + 0);
  __bf16* wkb = (__bf16*)(wp + 1 * 131072);
  __bf16* wvb = (__bf16*)(wp + 2 * 131072);
  __bf16* wmb = (__bf16*)(wp + 3 * 131072);
  __bf16* w1b = (__bf16*)(wp + 4 * 131072);                 // 512*512*2 = 524288
  __bf16* w2b = (__bf16*)(wp + 4 * 131072 + 524288);        // 256*512*2 = 262144
  float*  KV   = (float*)(wp + 4 * 131072 + 524288 + 262144);
  float*  Ksum = (float*)((char*)KV + (size_t)B_ * H_ * HD_ * HD_ * 4);
  const size_t kvBytes = (size_t)B_ * H_ * HD_ * HD_ * 4 + (size_t)B_ * H_ * HD_ * 4;

  // 1) convert inputs + weights to bf16; zero KV/Ksum accumulators
  cvt_f32_bf16<<<4096, 256, 0, stream>>>(query,   qbf, (long)M_ * D_);
  cvt_f32_bf16<<<4096, 256, 0, stream>>>(context, cbf, (long)M_ * D_);
  cvt_f32_bf16<<<256, 256, 0, stream>>>(Wq, wqb, (long)D_ * D_);
  cvt_f32_bf16<<<256, 256, 0, stream>>>(Wk, wkb, (long)D_ * D_);
  cvt_f32_bf16<<<256, 256, 0, stream>>>(Wv, wvb, (long)D_ * D_);
  cvt_f32_bf16<<<256, 256, 0, stream>>>(Wm, wmb, (long)D_ * D_);
  cvt_f32_bf16<<<512, 256, 0, stream>>>(W1, w1b, (long)D2_ * D2_);
  cvt_f32_bf16<<<512, 256, 0, stream>>>(W2, w2b, (long)D_ * D2_);
  hipMemsetAsync(KV, 0, kvBytes, stream);

  // 2) projections with fused phi / scaling (block tile 64x256 -> grid.y = N/256)
  dim3 gD(M_ / 64, D_ / 256);
  gemm_bf16_act<<<gD, 256, 0, stream>>>(qbf, qbf, D_, D_, wqb, D_, D_,
                                        nullptr, Qf, D_, ACT_ELU1, 0.f);
  gemm_bf16_act<<<gD, 256, 0, stream>>>(cbf, cbf, D_, D_, wkb, D_, D_,
                                        nullptr, Kf, D_, ACT_ELU1, 0.f);
  gemm_bf16_act<<<gD, 256, 0, stream>>>(cbf, cbf, D_, D_, wvb, D_, D_,
                                        nullptr, Vs, D_, ACT_SCALE, 1.f / (float)S_);

  // 3) per-head KV and Ksum reductions (s-split + f32 atomics)
  dim3 gKV(B_ * H_, KV_SPLIT);
  kv_ksum_kernel<<<gKV, 256, 0, stream>>>(Kf, Vs, KV, Ksum);

  // 4) message = Qf @ KV * Z * S  -> bf16
  message_kernel<<<M_, 256, 0, stream>>>(Qf, KV, Ksum, msg);

  // 5) merge heads: mergeF = msg @ Wmerge^T (f32)
  gemm_bf16_act<<<gD, 256, 0, stream>>>(msg, msg, D_, D_, wmb, D_, D_,
                                        mergeF, nullptr, D_, ACT_F32, 0.f);

  // 6) LayerNorm1 -> bf16
  ln_kernel<<<M_, 256, 0, stream>>>(mergeF, g1, b1, nullptr, nullptr, msg_ln);

  // 7) MLP1: mid = relu(concat(qbf, msg_ln) @ W1^T), K=512 via split-A
  dim3 gM1(M_ / 64, D2_ / 256);
  gemm_bf16_act<<<gM1, 256, 0, stream>>>(qbf, msg_ln, D_, D_, w1b, D2_, D2_,
                                         nullptr, mid, D2_, ACT_RELU, 0.f);

  // 8) MLP2: o2 = mid @ W2^T (f32)
  gemm_bf16_act<<<gD, 256, 0, stream>>>(mid, mid, D2_, D2_, w2b, D2_, D2_,
                                        o2, nullptr, D_, ACT_F32, 0.f);

  // 9) out = query + LayerNorm2(o2)
  ln_kernel<<<M_, 256, 0, stream>>>(o2, g2, b2, query, (float*)d_out, nullptr);
}
